// RCAN_56564719288342
// MI455X (gfx1250) — 8-bit matrix operand order repaired and checked by reading, compiled, not run on hardware
//
#include <hip/hip_runtime.h>
#include <hip/hip_bf16.h>
#include <math.h>
#include <stdint.h>


typedef int v8i __attribute__((ext_vector_type(8)));

#define HW      9216
#define T_ELEMS (4 * 64 * HW)
#define NPIX    (4 * HW)

__device__ __forceinline__ float leakyf(float v, float s) { return v >= 0.f ? v : s * v; }

__global__ void head_kernel(const float* __restrict__ x, const float* __restrict__ w,
                            const float* __restrict__ bias, float* __restrict__ h,
                            float* __restrict__ res) {
  int t = blockIdx.x * blockDim.x + threadIdx.x;
  if (t >= T_ELEMS) return;
  int rem = t % HW;
  int co  = (t / HW) & 63;
  int b   = t / (64 * HW);
  int yy = rem / 96, xx = rem % 96;
  const float m[3] = {114.444f, 111.4605f, 103.02f};
  float acc = bias[co];
  for (int ci = 0; ci < 3; ++ci)
    for (int kh = 0; kh < 3; ++kh) {
      int sy = yy + kh - 1;
      if (sy < 0 || sy >= 96) continue;
      for (int kw = 0; kw < 3; ++kw) {
        int sx = xx + kw - 1;
        if (sx < 0 || sx >= 96) continue;
        acc += (x[(b * 3 + ci) * HW + sy * 96 + sx] - m[ci]) *
               w[((co * 3 + ci) * 3 + kh) * 3 + kw];
      }
    }
  h[t] = acc;
  res[t] = acc;
}

__global__ void signpack_kernel(const float* __restrict__ y, unsigned int* __restrict__ sx) {
  int t = blockIdx.x * blockDim.x + threadIdx.x;
  if (t >= NPIX * 16) return;
  int p = t >> 4, q = t & 15;
  int b = p / HW, rem = p % HW;
  unsigned int pk = 0;
  for (int i = 0; i < 4; ++i) {
    int c = q * 4 + i;
    float v = y[(size_t)(b * 64 + c) * HW + rem];
    unsigned int s = v > 0.f ? 0x01u : (v < 0.f ? 0xFFu : 0x00u);
    pk |= s << (8 * i);
  }
  sx[t] = pk;
}

__global__ void packw_kernel(const float* __restrict__ w, int* __restrict__ apack) {
  int t = blockIdx.x * blockDim.x + threadIdx.x;
  if (t >= 4 * 9 * 32) return;
  int lane = t & 31, pos = (t >> 5) % 9, mt = t / (9 * 32);
  int half = lane >> 4, co = mt * 16 + (lane & 15);
  for (int j = 0; j < 8; ++j) {
    int kb = ((j >= 4) ? 32 : 0) + ((j & 3) * 4) + half * 16;
    unsigned int pk = 0;
    for (int bb = 0; bb < 4; ++bb) {
      int ci = kb + bb;
      float wv = w[(size_t)(co * 64 + ci) * 9 + pos];
      unsigned int s = wv > 0.f ? 0x01u : (wv < 0.f ? 0xFFu : 0x00u);
      pk |= s << (8 * bb);
    }
    apack[t * 8 + j] = (int)pk;
  }
}

__global__ void __launch_bounds__(256)
binconv_wmma_kernel(const int* __restrict__ apack, const unsigned char* __restrict__ sx,
                    const float* __restrict__ bias,
                    const float* __restrict__ bng, const float* __restrict__ bnb,
                    const float* __restrict__ bnm, const float* __restrict__ bnv,
                    const float* add0, float* out0, const float* add1, float* out1,
                    int mode) {
  int wave = (blockIdx.x * blockDim.x + threadIdx.x) >> 5;
  int lane = threadIdx.x & 31;
  int mt = wave & 3;
  int nt = wave >> 2;
  int p0 = nt * 64;
  int b = p0 / HW, rem0 = p0 % HW;
  int col = lane & 15, half = lane >> 4;

  int syy[4], sxx[4];
#pragma unroll
  for (int s = 0; s < 4; ++s) {
    int rs = rem0 + s * 16;
    syy[s] = rs / 96;
    sxx[s] = (rs % 96) + col;
  }

  v8i acc[4];
#pragma unroll
  for (int s = 0; s < 4; ++s)
#pragma unroll
    for (int r = 0; r < 8; ++r) acc[s][r] = 0;

  const unsigned char* sxb = sx + (size_t)b * HW * 64 + half * 16;

  for (int kh = 0; kh < 3; ++kh) {
    for (int kw = 0; kw < 3; ++kw) {
      int pos = kh * 3 + kw;
      const int4* ap = reinterpret_cast<const int4*>(apack + ((mt * 9 + pos) * 32 + lane) * 8);
      int4 a0 = ap[0], a1 = ap[1];
      v8i av = {a0.x, a0.y, a0.z, a0.w, a1.x, a1.y, a1.z, a1.w};

#pragma unroll
      for (int s = 0; s < 4; ++s) {
        int sy = syy[s] + kh - 1;
        int sxp = sxx[s] + kw - 1;
        int4 b0 = {0, 0, 0, 0}, b1 = {0, 0, 0, 0};
        if (sy >= 0 && sy < 96 && sxp >= 0 && sxp < 96) {
          const unsigned char* sp = sxb + (size_t)(sy * 96 + sxp) * 64;
          b0 = *reinterpret_cast<const int4*>(sp);
          b1 = *reinterpret_cast<const int4*>(sp + 32);
        }
        v8i bv = {b0.x, b0.y, b0.z, b0.w, b1.x, b1.y, b1.z, b1.w};
        acc[s] = __builtin_amdgcn_wmma_i32_16x16x64_iu8(true, av, true, bv, acc[s],
                                                        false, false);
      }
    }
  }

#pragma unroll
  for (int r = 0; r < 8; ++r) {
    int co = mt * 16 + r + half * 8;
    float bi = bias ? bias[co] : 0.f;
    float inv = bng[co] * rsqrtf(bnv[co] + 1e-5f);
    float sh = bnb[co] - bnm[co] * inv;
    size_t chbase = (size_t)(b * 64 + co) * HW;
#pragma unroll
    for (int s = 0; s < 4; ++s) {
      float f = ((float)acc[s][r] + bi) * inv + sh;
      size_t idx = chbase + (size_t)(rem0 + s * 16 + col);
      if (mode == 0) {
        out0[idx] = leakyf(f + add0[idx], 0.1f);
      } else if (mode == 1) {
        out0[idx] = f + add0[idx];
      } else {
        float yn = f + add0[idx];
        out0[idx] = yn;
        out1[idx] = yn + add1[idx];
      }
    }
  }
}

__global__ void careduce_kernel(const float* __restrict__ t, float* __restrict__ mean) {
  __shared__ float red[256];
  int bc = blockIdx.x;
  const float* base = t + (size_t)bc * HW;
  float s = 0.f;
  for (int i = threadIdx.x; i < HW; i += 256) s += base[i];
  red[threadIdx.x] = s;
  __syncthreads();
  for (int off = 128; off > 0; off >>= 1) {
    if (threadIdx.x < off) red[threadIdx.x] += red[threadIdx.x + off];
    __syncthreads();
  }
  if (threadIdx.x == 0) mean[bc] = red[0] * (1.f / HW);
}

__global__ void camlp_kernel(const float* __restrict__ mean,
                             const float* __restrict__ w1, const float* __restrict__ b1,
                             const float* __restrict__ w2, const float* __restrict__ b2,
                             float* __restrict__ scale) {
  __shared__ float y1[4];
  int b = blockIdx.x, tid = threadIdx.x;
  if (tid < 4) {
    float s = b1[tid];
    for (int c = 0; c < 64; ++c) s += mean[b * 64 + c] * w1[tid * 64 + c];
    y1[tid] = leakyf(s, 0.01f);
  }
  __syncthreads();
  float s2 = b2[tid];
  for (int j = 0; j < 4; ++j) s2 += y1[j] * w2[tid * 4 + j];
  scale[b * 64 + tid] = 1.f / (1.f + expf(-s2));
}

__global__ void scale_kernel(const float* __restrict__ src, const float* __restrict__ scale,
                             float* __restrict__ dst) {
  int t = blockIdx.x * blockDim.x + threadIdx.x;
  if (t >= T_ELEMS) return;
  dst[t] = src[t] * scale[t / HW];
}

__global__ void tail_kernel(const float* __restrict__ res, const float* __restrict__ h,
                            const float* __restrict__ w, const float* __restrict__ bias,
                            float* __restrict__ out) {
  int t = blockIdx.x * blockDim.x + threadIdx.x;
  if (t >= 4 * 12 * HW) return;
  int rem = t % HW;
  int ct  = (t / HW) % 12;
  int b   = t / (12 * HW);
  int yy = rem / 96, xx = rem % 96;
  float acc = bias[ct];
  for (int ci = 0; ci < 64; ++ci)
    for (int kh = 0; kh < 3; ++kh) {
      int sy = yy + kh - 1;
      if (sy < 0 || sy >= 96) continue;
      for (int kw = 0; kw < 3; ++kw) {
        int sxp = xx + kw - 1;
        if (sxp < 0 || sxp >= 96) continue;
        size_t idx = (size_t)(b * 64 + ci) * HW + sy * 96 + sxp;
        acc += (res[idx] + h[idx]) * w[((ct * 64 + ci) * 3 + kh) * 3 + kw];
      }
    }
  int c = ct >> 2, ry = (ct >> 1) & 1, rx = ct & 1;
  const float m[3] = {114.444f, 111.4605f, 103.02f};
  out[((size_t)(b * 3 + c) * 192 + (2 * yy + ry)) * 192 + (2 * xx + rx)] = acc + m[c];
}

extern "C" void kernel_launch(void* const* d_in, const int* in_sizes, int n_in,
                              void* d_out, int out_size, void* d_ws, size_t ws_size,
                              hipStream_t stream) {
  (void)in_sizes; (void)n_in; (void)out_size; (void)ws_size;

  const float* X = (const float*)d_in[0];
  int i = 1;
  const float *pb1[3][4], *pb2[3][4], *pbn1[3][4][4], *pbn2[3][4][4];
  const float *pcab1[3][4], *pcab2[3][4], *pcaw1[3][4], *pcaw2[3][4], *pw1[3][4], *pw2[3][4];
  const float *ptbn[3][4], *ptw[3];
  for (int g = 0; g < 3; ++g) {
    for (int k = 0; k < 4; ++k) {
      pb1[g][k] = (const float*)d_in[i++];
      pb2[g][k] = (const float*)d_in[i++];
      for (int t = 0; t < 4; ++t) pbn1[g][k][t] = (const float*)d_in[i++];
      for (int t = 0; t < 4; ++t) pbn2[g][k][t] = (const float*)d_in[i++];
      pcab1[g][k] = (const float*)d_in[i++];
      pcab2[g][k] = (const float*)d_in[i++];
      pcaw1[g][k] = (const float*)d_in[i++];
      pcaw2[g][k] = (const float*)d_in[i++];
      pw1[g][k] = (const float*)d_in[i++];
      pw2[g][k] = (const float*)d_in[i++];
    }
    for (int t = 0; t < 4; ++t) ptbn[g][t] = (const float*)d_in[i++];
    ptw[g] = (const float*)d_in[i++];
  }
  const float* head_b = (const float*)d_in[i++];
  const float* head_w = (const float*)d_in[i++];
  const float* tail_b = (const float*)d_in[i++];
  const float* tail_w = (const float*)d_in[i++];

  uintptr_t p = (uintptr_t)d_ws;
  auto bump = [&](size_t bytes) -> void* {
    uintptr_t q = (p + 255) & ~(uintptr_t)255;
    p = q + bytes;
    return (void*)q;
  };
  size_t TB = (size_t)T_ELEMS * 4;
  float* h   = (float*)bump(TB);
  float* res = (float*)bump(TB);
  float* y   = (float*)bump(TB);
  float* o1  = (float*)bump(TB);
  float* o2  = (float*)bump(TB);
  unsigned char* sx = (unsigned char*)bump(T_ELEMS);
  int* apack = (int*)bump((size_t)4 * 9 * 32 * 8 * 4);
  float* camean  = (float*)bump(256 * 4);
  float* cascale = (float*)bump(256 * 4);

  const int BC_BLOCKS = 288;

  head_kernel<<<T_ELEMS / 256, 256, 0, stream>>>(X, head_w, head_b, h, res);

  for (int g = 0; g < 3; ++g) {
    hipMemcpyAsync(y, res, TB, hipMemcpyDeviceToDevice, stream);
    for (int k = 0; k < 4; ++k) {
      signpack_kernel<<<(NPIX * 16) / 256, 256, 0, stream>>>(y, (unsigned int*)sx);
      packw_kernel<<<5, 256, 0, stream>>>(pw1[g][k], apack);
      binconv_wmma_kernel<<<BC_BLOCKS, 256, 0, stream>>>(
          apack, sx, pb1[g][k], pbn1[g][k][0], pbn1[g][k][1], pbn1[g][k][2], pbn1[g][k][3],
          y, o1, nullptr, nullptr, 0);
      packw_kernel<<<5, 256, 0, stream>>>(pw2[g][k], apack);
      binconv_wmma_kernel<<<BC_BLOCKS, 256, 0, stream>>>(
          apack, sx, pb2[g][k], pbn2[g][k][0], pbn2[g][k][1], pbn2[g][k][2], pbn2[g][k][3],
          o1, o2, nullptr, nullptr, 1);
      careduce_kernel<<<256, 256, 0, stream>>>(o2, camean);
      camlp_kernel<<<4, 64, 0, stream>>>(camean, pcaw1[g][k], pcab1[g][k],
                                         pcaw2[g][k], pcab2[g][k], cascale);
      scale_kernel<<<T_ELEMS / 256, 256, 0, stream>>>(o2, cascale, y);
    }
    signpack_kernel<<<(NPIX * 16) / 256, 256, 0, stream>>>(y, (unsigned int*)sx);
    packw_kernel<<<5, 256, 0, stream>>>(ptw[g], apack);
    binconv_wmma_kernel<<<BC_BLOCKS, 256, 0, stream>>>(
        apack, sx, nullptr, ptbn[g][0], ptbn[g][1], ptbn[g][2], ptbn[g][3],
        y, y, res, res, 2);
  }

  tail_kernel<<<1728, 256, 0, stream>>>(res, h, tail_w, tail_b, (float*)d_out);
}
